// CombinedAttention_90314572300388
// MI455X (gfx1250) — compile-verified, matrix operand order checked by reading, not run on hardware
//
#include <hip/hip_runtime.h>
#include <hip/hip_bf16.h>
#include <math.h>

// ---------------------------------------------------------------------------
// Problem constants: B=8, C=256, CQ=32, H=96, W=96
// ---------------------------------------------------------------------------
#define BB   8
#define CC   256
#define CQQ  32
#define HH   96
#define WW   96
#define HWW  (HH * WW)          // 9216
#define NPIX (BB * HWW)         // 73728
#define NTOT (BB * CC * HWW)    // 18,874,368
#define BN_EPS 1e-5f

// Workspace layout (byte offsets, 256-aligned). Lifetimes overlapped.
#define B_XT   0ul                        // x NHWC bf16 ; later vt
#define B_MS   37748736ul                 // ms NHWC bf16; later att_bf
#define B_QT   75497472ul                 // q NHWC bf16 ; later pooled
#define B_KT   80216064ul                 // k NHWC bf16 ; later sescale
#define B_VBF  84934656ul                 // v NCHW bf16
#define B_E    122683392ul                // scores f32  ; later xrt bf16
#define B_OSUM 179306496ul                // outH+outW f32; later x1 f32
#define B_XR   254803968ul                // xr f32
#define B_WR1  330301440ul                // repacked 3x3 weights bf16
#define B_WR2  (B_WR1 + 1179648ul)
#define B_WR3  (B_WR2 + 1179648ul)
#define B_W3R  (B_WR3 + 1179648ul)
#define B_WQ   (B_W3R + 1179648ul)
#define B_WK   (B_WQ + 16384ul)
#define B_WV   (B_WK + 16384ul)
#define B_W1   (B_WV + 131072ul)

// ---------------------------------------------------------------------------
// WMMA fragments: A/B assembled from two b128 loads per lane
// (K-groups [klo, klo+7], [klo+16, klo+23] per CDNA5 16-bit layout).
// ---------------------------------------------------------------------------
typedef __attribute__((ext_vector_type(16))) __bf16 v16bf;
typedef __attribute__((ext_vector_type(8)))  float  v8f;

union FragAB { v16bf b; struct { uint4 lo, hi; } q; };
union FragC  { v8f v; float e[8]; };
union Pack8  { uint4 q; unsigned short s[8]; };

__device__ __forceinline__ unsigned short f2bf(float f) {
    union { float f; unsigned u; } cv; cv.f = f;
    unsigned u = cv.u;
    return (unsigned short)((u + 0x7FFFu + ((u >> 16) & 1u)) >> 16);  // RNE
}

#define LANE_DECODE int lane = threadIdx.x & 31; int half = lane >> 4; \
                    int mn = lane & 15; int klo = half * 8;

__device__ __forceinline__ void ldfrag(FragAB& f, const unsigned short* p, int klo) {
    f.q.lo = *(const uint4*)(p + klo);
    f.q.hi = *(const uint4*)(p + klo + 16);
}
__device__ __forceinline__ void zfrag(FragAB& f) {
    f.q.lo = make_uint4(0, 0, 0, 0);
    f.q.hi = make_uint4(0, 0, 0, 0);
}
#define WMMA_BF16(acc, A, Bm) \
    acc.v = __builtin_amdgcn_wmma_f32_16x16x32_bf16(false, (A).b, false, (Bm).b, \
                                                    (short)0, acc.v, false, false)

// ---------------------------------------------------------------------------
// Conversion / repack kernels
// ---------------------------------------------------------------------------
// f32 NCHW -> bf16 NHWC (x and xr). 32(w) x 32(c) LDS tile, both sides coalesced.
__global__ __launch_bounds__(256)
void k_nchw2nhwc(const float* __restrict__ in, unsigned short* __restrict__ out) {
    __shared__ float tile[32][33];
    unsigned bid = blockIdx.x;
    int ct = bid & 7;      unsigned t = bid >> 3;
    int wt = t % 3;        t /= 3;
    int h  = t % HH;       int b = t / HH;
    int c0 = ct * 32, w0 = wt * 32;
    int tx = threadIdx.x & 31, ty = threadIdx.x >> 5;
#pragma unroll
    for (int i = 0; i < 4; ++i) {
        int cl = ty + i * 8;
        tile[cl][tx] = in[((long)(b * CC + c0 + cl) * HH + h) * WW + w0 + tx];
    }
    __syncthreads();
#pragma unroll
    for (int i = 0; i < 4; ++i) {
        int wl = ty + i * 8;
        out[((long)(b * HH + h) * WW + w0 + wl) * CC + c0 + tx] = f2bf(tile[tx][wl]);
    }
}

// OIHW f32 3x3 weights -> [co][tap][ci] bf16 (K-contiguous A operand)
__global__ void k_repack3x3(const float* __restrict__ in, unsigned short* __restrict__ out) {
    int i = blockIdx.x * blockDim.x + threadIdx.x;
    if (i >= CC * 9 * CC) return;
    int ci = i & 255, tap = (i >> 8) % 9, co = i / (9 * CC);
    out[i] = f2bf(in[(long)(co * CC + ci) * 9 + tap]);
}

__global__ void k_cvtbf(const float* __restrict__ in, unsigned short* __restrict__ out, int n) {
    int i = blockIdx.x * blockDim.x + threadIdx.x;
    if (i < n) out[i] = f2bf(in[i]);
}

// bf16 v NCHW [b][c][H][w] -> vt [b][w][c][H]
__global__ __launch_bounds__(256)
void k_vtrans(const unsigned short* __restrict__ vin, unsigned short* __restrict__ vt) {
    __shared__ unsigned short tile[32][33];
    unsigned bid = blockIdx.x;
    int Ht = bid % 3;      unsigned t = bid / 3;
    int wt = t % 3;        t /= 3;
    int c  = t % CC;       int b = t / CC;
    int H0 = Ht * 32, w0 = wt * 32;
    int tx = threadIdx.x & 31, ty = threadIdx.x >> 5;
#pragma unroll
    for (int i = 0; i < 4; ++i) {
        int hl = ty + i * 8;
        tile[hl][tx] = vin[(long)(b * CC + c) * HWW + (H0 + hl) * WW + w0 + tx];
    }
    __syncthreads();
#pragma unroll
    for (int i = 0; i < 4; ++i) {
        int wl = ty + i * 8;
        vt[((long)(b * WW + w0 + wl) * CC + c) * HH + H0 + tx] = tile[tx][wl];
    }
}

// ---------------------------------------------------------------------------
// Multi-scale conv: ms = sum of three dilated 3x3 convs.
// 4-way co register blocking: each B fragment feeds 4 WMMAs.
// ---------------------------------------------------------------------------
__global__ __launch_bounds__(32)
void k_msconv(const unsigned short* __restrict__ xt,
              const unsigned short* __restrict__ wr1,
              const unsigned short* __restrict__ wr2,
              const unsigned short* __restrict__ wr3,
              const float* __restrict__ bm1, const float* __restrict__ bm2,
              const float* __restrict__ bm3,
              unsigned short* __restrict__ msbf) {
    LANE_DECODE;
    unsigned bid = blockIdx.x;
    int grp = bid & 3;    unsigned t = bid >> 2;
    int wt = t % 6;       t /= 6;
    int h  = t % HH;      int b = t / HH;
    int w0 = wt * 16;
    int co_base = grp * 64;

    FragC acc[4];
#pragma unroll
    for (int j = 0; j < 4; ++j)
#pragma unroll
        for (int i = 0; i < 8; ++i) acc[j].e[i] = 0.0f;

    for (int cv = 0; cv < 3; ++cv) {
        const unsigned short* Wr = (cv == 0) ? wr1 : (cv == 1) ? wr2 : wr3;
        int dil = cv + 1;
        for (int kh = 0; kh < 3; ++kh) {
            int ih = h + dil * (kh - 1);
            if (ih < 0 || ih >= HH) continue;
            for (int kw = 0; kw < 3; ++kw) {
                int iw   = w0 + mn + dil * (kw - 1);
                bool okw = (iw >= 0) && (iw < WW);
                int tap  = kh * 3 + kw;
                const unsigned short* bp = xt + ((long)(b * HH + ih) * WW + iw) * CC;
                for (int ch = 0; ch < 8; ++ch) {
                    int ci0 = ch * 32;
                    if (ch < 7) __builtin_prefetch(bp + ci0 + 32, 0, 0);
                    FragAB Bm;
                    if (okw) ldfrag(Bm, bp + ci0, klo); else zfrag(Bm);
#pragma unroll
                    for (int j = 0; j < 4; ++j) {
                        FragAB A;
                        ldfrag(A, Wr + ((long)(co_base + j * 16 + mn) * 9 + tap) * CC + ci0, klo);
                        WMMA_BF16(acc[j], A, Bm);
                    }
                }
            }
        }
    }
#pragma unroll
    for (int j = 0; j < 4; ++j) {
        Pack8 o;
#pragma unroll
        for (int r = 0; r < 8; ++r) {
            int co = co_base + j * 16 + half * 8 + r;
            o.s[r] = f2bf(acc[j].e[r] + bm1[co] + bm2[co] + bm3[co]);
        }
        *(uint4*)(msbf + ((long)(b * HH + h) * WW + w0 + mn) * CC
                       + co_base + j * 16 + half * 8) = o.q;
    }
}

// ---------------------------------------------------------------------------
// 1x1 GEMMs (NHWC bf16 inputs, bf16 weight rows).
// ---------------------------------------------------------------------------
// q/k: Cout=32, both co tiles in one wave (B fragment reused 2x). NHWC bf16 out.
__global__ __launch_bounds__(32)
void k_qkgemm(const unsigned short* __restrict__ in, const unsigned short* __restrict__ Wbf,
              const float* __restrict__ bias, unsigned short* __restrict__ out) {
    LANE_DECODE;
    unsigned pt = blockIdx.x;
    long pix = (long)(pt / (HWW / 16)) * HWW + (pt % (HWW / 16)) * 16 + mn;

    FragC acc[2];
#pragma unroll
    for (int j = 0; j < 2; ++j)
#pragma unroll
        for (int i = 0; i < 8; ++i) acc[j].e[i] = 0.0f;
    const unsigned short* bp = in + pix * CC;
    for (int ch = 0; ch < 8; ++ch) {
        FragAB Bm;
        ldfrag(Bm, bp + ch * 32, klo);
#pragma unroll
        for (int j = 0; j < 2; ++j) {
            FragAB A;
            ldfrag(A, Wbf + (long)(j * 16 + mn) * CC + ch * 32, klo);
            WMMA_BF16(acc[j], A, Bm);
        }
    }
#pragma unroll
    for (int j = 0; j < 2; ++j) {
        Pack8 o;
#pragma unroll
        for (int r = 0; r < 8; ++r)
            o.s[r] = f2bf(acc[j].e[r] + bias[j * 16 + half * 8 + r]);
        *(uint4*)(out + pix * CQQ + j * 16 + half * 8) = o.q;
    }
}

// v: Cout=256, 4-way co blocking, NCHW bf16 output.
__global__ __launch_bounds__(32)
void k_vgemm(const unsigned short* __restrict__ in, const unsigned short* __restrict__ Wbf,
             const float* __restrict__ bias, unsigned short* __restrict__ out) {
    LANE_DECODE;
    unsigned bid = blockIdx.x;
    int grp = bid & 3;
    unsigned pt = bid >> 2;
    int pb = pt / (HWW / 16);
    int hw0 = (pt % (HWW / 16)) * 16;
    int co_base = grp * 64;

    FragC acc[4];
#pragma unroll
    for (int j = 0; j < 4; ++j)
#pragma unroll
        for (int i = 0; i < 8; ++i) acc[j].e[i] = 0.0f;
    const unsigned short* bp = in + ((long)pb * HWW + hw0 + mn) * CC;
    for (int ch = 0; ch < 8; ++ch) {
        FragAB Bm;
        ldfrag(Bm, bp + ch * 32, klo);
#pragma unroll
        for (int j = 0; j < 4; ++j) {
            FragAB A;
            ldfrag(A, Wbf + (long)(co_base + j * 16 + mn) * CC + ch * 32, klo);
            WMMA_BF16(acc[j], A, Bm);
        }
    }
#pragma unroll
    for (int j = 0; j < 4; ++j)
#pragma unroll
        for (int r = 0; r < 8; ++r) {
            int co = co_base + j * 16 + half * 8 + r;
            out[(long)(pb * CC + co) * HWW + hw0 + mn] = f2bf(acc[j].e[r] + bias[co]);
        }
}

// x1: Cout=256, 4-way co blocking, fused BN, f32 NCHW output.
__global__ __launch_bounds__(32)
void k_x1gemm(const unsigned short* __restrict__ in, const unsigned short* __restrict__ Wbf,
              const float* __restrict__ bias, float* __restrict__ out,
              const float* __restrict__ bng, const float* __restrict__ bnb,
              const float* __restrict__ bnm, const float* __restrict__ bnv) {
    LANE_DECODE;
    unsigned bid = blockIdx.x;
    int grp = bid & 3;
    unsigned pt = bid >> 2;
    int pb = pt / (HWW / 16);
    int hw0 = (pt % (HWW / 16)) * 16;
    int co_base = grp * 64;

    FragC acc[4];
#pragma unroll
    for (int j = 0; j < 4; ++j)
#pragma unroll
        for (int i = 0; i < 8; ++i) acc[j].e[i] = 0.0f;
    const unsigned short* bp = in + ((long)pb * HWW + hw0 + mn) * CC;
    for (int ch = 0; ch < 8; ++ch) {
        FragAB Bm;
        ldfrag(Bm, bp + ch * 32, klo);
#pragma unroll
        for (int j = 0; j < 4; ++j) {
            FragAB A;
            ldfrag(A, Wbf + (long)(co_base + j * 16 + mn) * CC + ch * 32, klo);
            WMMA_BF16(acc[j], A, Bm);
        }
    }
#pragma unroll
    for (int j = 0; j < 4; ++j)
#pragma unroll
        for (int r = 0; r < 8; ++r) {
            int co = co_base + j * 16 + half * 8 + r;
            float s = bng[co] * rsqrtf(bnv[co] + BN_EPS);
            out[(long)(pb * CC + co) * HWW + hw0 + mn] =
                (acc[j].e[r] + bias[co]) * s + bnb[co] - bnm[co] * s;
        }
}

// ---------------------------------------------------------------------------
// Attention scores (K = CQ = 32: one WMMA per tile). The q A-fragment is
// loaded once per wave and reused across all 6 key tiles.
// ---------------------------------------------------------------------------
__global__ __launch_bounds__(32)
void k_scoreH(const unsigned short* __restrict__ qt, const unsigned short* __restrict__ kt,
              float* __restrict__ e) {
    LANE_DECODE;
    unsigned bid = blockIdx.x;
    int ht = bid % 6;  unsigned t = bid / 6;
    int w  = t % WW;   int b = t / WW;

    FragAB A;
    ldfrag(A, qt + ((long)(b * HH + ht * 16 + mn) * WW + w) * CQQ, klo);
    for (int Ht = 0; Ht < 6; ++Ht) {
        FragAB Bm;
        ldfrag(Bm, kt + ((long)(b * HH + Ht * 16 + mn) * WW + w) * CQQ, klo);
        FragC acc;
#pragma unroll
        for (int i = 0; i < 8; ++i) acc.e[i] = 0.0f;
        WMMA_BF16(acc, A, Bm);
#pragma unroll
        for (int r = 0; r < 8; ++r) {
            int h  = ht * 16 + half * 8 + r;
            int Hc = Ht * 16 + mn;
            e[(((long)b * HH + h) * WW + w) * 192 + Hc] = (h == Hc) ? -INFINITY : acc.e[r];
        }
    }
}

__global__ __launch_bounds__(32)
void k_scoreW(const unsigned short* __restrict__ qt, const unsigned short* __restrict__ kt,
              float* __restrict__ e) {
    LANE_DECODE;
    unsigned bid = blockIdx.x;
    int wt = bid % 6;  unsigned t = bid / 6;
    int h  = t % HH;   int b = t / HH;

    FragAB A;
    ldfrag(A, qt + ((long)(b * HH + h) * WW + wt * 16 + mn) * CQQ, klo);
    for (int Wt = 0; Wt < 6; ++Wt) {
        FragAB Bm;
        ldfrag(Bm, kt + ((long)(b * HH + h) * WW + Wt * 16 + mn) * CQQ, klo);
        FragC acc;
#pragma unroll
        for (int i = 0; i < 8; ++i) acc.e[i] = 0.0f;
        WMMA_BF16(acc, A, Bm);
#pragma unroll
        for (int r = 0; r < 8; ++r)
            e[(((long)b * HH + h) * WW + wt * 16 + half * 8 + r) * 192
              + 96 + Wt * 16 + mn] = acc.e[r];
    }
}

// ---------------------------------------------------------------------------
// Softmax over 192 per (b,h,w); f32 scores in, bf16 probabilities out.
// ---------------------------------------------------------------------------
__global__ __launch_bounds__(32)
void k_softmax192(const float* __restrict__ e, unsigned short* __restrict__ att) {
    int lane = threadIdx.x & 31;
    const float* row = e + (long)blockIdx.x * 192;
    unsigned short* orow = att + (long)blockIdx.x * 192;
    float v[6], mx = -INFINITY;
#pragma unroll
    for (int j = 0; j < 6; ++j) { v[j] = row[lane + 32 * j]; mx = fmaxf(mx, v[j]); }
#pragma unroll
    for (int s = 16; s >= 1; s >>= 1) mx = fmaxf(mx, __shfl_xor(mx, s, 32));
    float sum = 0.0f;
#pragma unroll
    for (int j = 0; j < 6; ++j) { v[j] = __expf(v[j] - mx); sum += v[j]; }
#pragma unroll
    for (int s = 16; s >= 1; s >>= 1) sum += __shfl_xor(sum, s, 32);
    float inv = 1.0f / sum;
#pragma unroll
    for (int j = 0; j < 6; ++j) orow[lane + 32 * j] = f2bf(v[j] * inv);
}

// ---------------------------------------------------------------------------
// att x V, 4-way c blocking (B att fragment reused 4x). K=96: 3 chunks.
// ---------------------------------------------------------------------------
__global__ __launch_bounds__(32)
void k_outH(const unsigned short* __restrict__ vt, const unsigned short* __restrict__ att,
            float* __restrict__ out) {
    LANE_DECODE;
    unsigned bid = blockIdx.x;
    int ht = bid % 6;  unsigned t = bid / 6;
    int cg = t % 4;    t /= 4;
    int w  = t % WW;   int b = t / WW;
    int c_base = cg * 64;

    FragC acc[4];
#pragma unroll
    for (int j = 0; j < 4; ++j)
#pragma unroll
        for (int i = 0; i < 8; ++i) acc[j].e[i] = 0.0f;
    const unsigned short* bp = att + (((long)b * HH + ht * 16 + mn) * WW + w) * 192;
    for (int kc = 0; kc < 3; ++kc) {
        FragAB Bm;
        ldfrag(Bm, bp + kc * 32, klo);
#pragma unroll
        for (int j = 0; j < 4; ++j) {
            FragAB A;
            ldfrag(A, vt + ((long)(b * WW + w) * CC + c_base + j * 16 + mn) * HH + kc * 32, klo);
            WMMA_BF16(acc[j], A, Bm);
        }
    }
#pragma unroll
    for (int j = 0; j < 4; ++j)
#pragma unroll
        for (int r = 0; r < 8; ++r) {
            int c = c_base + j * 16 + half * 8 + r;
            out[(long)(b * CC + c) * HWW + (ht * 16 + mn) * WW + w] = acc[j].e[r];
        }
}

__global__ __launch_bounds__(32)
void k_outW(const unsigned short* __restrict__ vbf, const unsigned short* __restrict__ att,
            float* __restrict__ out) {
    LANE_DECODE;
    unsigned bid = blockIdx.x;
    int wt = bid % 6;  unsigned t = bid / 6;
    int cg = t % 4;    t /= 4;
    int h  = t % HH;   int b = t / HH;
    int c_base = cg * 64;

    FragC acc[4];
#pragma unroll
    for (int j = 0; j < 4; ++j)
#pragma unroll
        for (int i = 0; i < 8; ++i) acc[j].e[i] = 0.0f;
    const unsigned short* bp = att + (((long)b * HH + h) * WW + wt * 16 + mn) * 192 + 96;
    for (int kc = 0; kc < 3; ++kc) {
        FragAB Bm;
        ldfrag(Bm, bp + kc * 32, klo);
#pragma unroll
        for (int j = 0; j < 4; ++j) {
            FragAB A;
            ldfrag(A, vbf + (long)(b * CC + c_base + j * 16 + mn) * HWW + h * WW + kc * 32, klo);
            WMMA_BF16(acc[j], A, Bm);
        }
    }
#pragma unroll
    for (int j = 0; j < 4; ++j)
#pragma unroll
        for (int r = 0; r < 8; ++r) {
            int c = c_base + j * 16 + half * 8 + r;
            long idx = (long)(b * CC + c) * HWW + h * WW + wt * 16 + mn;
            out[idx] += acc[j].e[r];
        }
}

// ---------------------------------------------------------------------------
// Elementwise / reduction tail
// ---------------------------------------------------------------------------
__global__ void k_xr(const float* __restrict__ osum, const float* __restrict__ x,
                     const float* __restrict__ gamma, float* __restrict__ xr) {
    long i = (long)blockIdx.x * blockDim.x + threadIdx.x;
    if (i < NTOT) xr[i] = gamma[0] * osum[i] + x[i];
}

__global__ __launch_bounds__(256)
void k_pool(const float* __restrict__ xr, float* __restrict__ pooled) {
    __shared__ float sd[256];
    const float* p = xr + (long)blockIdx.x * HWW;
    float s = 0.0f;
    for (int i = threadIdx.x; i < HWW; i += 256) s += p[i];
    sd[threadIdx.x] = s; __syncthreads();
    for (int st = 128; st >= 1; st >>= 1) {
        if ((int)threadIdx.x < st) sd[threadIdx.x] += sd[threadIdx.x + st];
        __syncthreads();
    }
    if (threadIdx.x == 0) pooled[blockIdx.x] = sd[0] * (1.0f / HWW);
}

__global__ __launch_bounds__(32)
void k_se(const float* __restrict__ pooled, const float* __restrict__ wse,
          const float* __restrict__ bse, float* __restrict__ sescale) {
    int lane = threadIdx.x & 31;
    int b = blockIdx.x / CC, c = blockIdx.x % CC;
    float s = 0.0f;
    for (int ci = lane; ci < CC; ci += 32)
        s += wse[(long)c * CC + ci] * pooled[b * CC + ci];
#pragma unroll
    for (int st = 16; st >= 1; st >>= 1) s += __shfl_xor(s, st, 32);
    if (lane == 0) sescale[blockIdx.x] = 1.0f / (1.0f + __expf(-(s + bse[c])));
}

// ---------------------------------------------------------------------------
// Final: out = silu( BN(conv3x3(xr)) + x1 + sescale*xr ), 4-way co blocking.
// ---------------------------------------------------------------------------
__global__ __launch_bounds__(32)
void k_final(const unsigned short* __restrict__ xrt, const unsigned short* __restrict__ w3r,
             const float* __restrict__ b3,
             const float* __restrict__ bng, const float* __restrict__ bnb,
             const float* __restrict__ bnm, const float* __restrict__ bnv,
             const float* __restrict__ x1, const float* __restrict__ xr,
             const float* __restrict__ sescale, float* __restrict__ out) {
    LANE_DECODE;
    unsigned bid = blockIdx.x;
    int grp = bid & 3;    unsigned t = bid >> 2;
    int wt = t % 6;       t /= 6;
    int h  = t % HH;      int b = t / HH;
    int w0 = wt * 16;
    int co_base = grp * 64;

    FragC acc[4];
#pragma unroll
    for (int j = 0; j < 4; ++j)
#pragma unroll
        for (int i = 0; i < 8; ++i) acc[j].e[i] = 0.0f;

    for (int kh = 0; kh < 3; ++kh) {
        int ih = h + kh - 1;
        if (ih < 0 || ih >= HH) continue;
        for (int kw = 0; kw < 3; ++kw) {
            int iw   = w0 + mn + kw - 1;
            bool okw = (iw >= 0) && (iw < WW);
            int tap  = kh * 3 + kw;
            const unsigned short* bp = xrt + ((long)(b * HH + ih) * WW + iw) * CC;
            for (int ch = 0; ch < 8; ++ch) {
                if (ch < 7) __builtin_prefetch(bp + ch * 32 + 32, 0, 0);
                FragAB Bm;
                if (okw) ldfrag(Bm, bp + ch * 32, klo); else zfrag(Bm);
#pragma unroll
                for (int j = 0; j < 4; ++j) {
                    FragAB A;
                    ldfrag(A, w3r + ((long)(co_base + j * 16 + mn) * 9 + tap) * CC + ch * 32, klo);
                    WMMA_BF16(acc[j], A, Bm);
                }
            }
        }
    }
#pragma unroll
    for (int j = 0; j < 4; ++j)
#pragma unroll
        for (int r = 0; r < 8; ++r) {
            int co = co_base + j * 16 + half * 8 + r;
            long idx = (long)(b * CC + co) * HWW + h * WW + w0 + mn;
            float s  = bng[co] * rsqrtf(bnv[co] + BN_EPS);
            float x2 = (acc[j].e[r] + b3[co]) * s + bnb[co] - bnm[co] * s;
            float z  = x1[idx] + x2 + sescale[b * CC + co] * xr[idx];
            out[idx] = z / (1.0f + __expf(-z));   // SiLU
        }
}

// ---------------------------------------------------------------------------
// Host launcher
// ---------------------------------------------------------------------------
extern "C" void kernel_launch(void* const* d_in, const int* in_sizes, int n_in,
                              void* d_out, int out_size, void* d_ws, size_t ws_size,
                              hipStream_t stream) {
    (void)in_sizes; (void)n_in; (void)out_size; (void)ws_size;
    const float* x     = (const float*)d_in[0];
    const float* w_ms1 = (const float*)d_in[1];  const float* b_ms1 = (const float*)d_in[2];
    const float* w_ms2 = (const float*)d_in[3];  const float* b_ms2 = (const float*)d_in[4];
    const float* w_ms3 = (const float*)d_in[5];  const float* b_ms3 = (const float*)d_in[6];
    const float* wq = (const float*)d_in[7];     const float* bq = (const float*)d_in[8];
    const float* wk = (const float*)d_in[9];     const float* bk = (const float*)d_in[10];
    const float* wv = (const float*)d_in[11];    const float* bv = (const float*)d_in[12];
    const float* gamma = (const float*)d_in[13];
    const float* w1 = (const float*)d_in[14];    const float* b1 = (const float*)d_in[15];
    const float* bn1g = (const float*)d_in[16];  const float* bn1b = (const float*)d_in[17];
    const float* bn1m = (const float*)d_in[18];  const float* bn1v = (const float*)d_in[19];
    const float* w3 = (const float*)d_in[20];    const float* b3 = (const float*)d_in[21];
    const float* bn3g = (const float*)d_in[22];  const float* bn3b = (const float*)d_in[23];
    const float* bn3m = (const float*)d_in[24];  const float* bn3v = (const float*)d_in[25];
    const float* w_se = (const float*)d_in[26];  const float* b_se = (const float*)d_in[27];

    char* ws = (char*)d_ws;
    unsigned short* xt   = (unsigned short*)(ws + B_XT);
    unsigned short* vt   = (unsigned short*)(ws + B_XT);    // reuse after msconv
    unsigned short* msbf = (unsigned short*)(ws + B_MS);
    unsigned short* att  = (unsigned short*)(ws + B_MS);    // reuse after qkv
    unsigned short* qt   = (unsigned short*)(ws + B_QT);
    unsigned short* kt   = (unsigned short*)(ws + B_KT);
    unsigned short* vbf  = (unsigned short*)(ws + B_VBF);
    float*          e    = (float*)(ws + B_E);
    unsigned short* xrt  = (unsigned short*)(ws + B_E);     // reuse after softmax
    float*          osum = (float*)(ws + B_OSUM);
    float*          x1   = (float*)(ws + B_OSUM);           // reuse after xr
    float*          xr   = (float*)(ws + B_XR);
    unsigned short* wr1  = (unsigned short*)(ws + B_WR1);
    unsigned short* wr2  = (unsigned short*)(ws + B_WR2);
    unsigned short* wr3  = (unsigned short*)(ws + B_WR3);
    unsigned short* w3r  = (unsigned short*)(ws + B_W3R);
    unsigned short* wqb  = (unsigned short*)(ws + B_WQ);
    unsigned short* wkb  = (unsigned short*)(ws + B_WK);
    unsigned short* wvb  = (unsigned short*)(ws + B_WV);
    unsigned short* w1b  = (unsigned short*)(ws + B_W1);
    float* pooled = (float*)(ws + B_QT);                    // reuse after scores
    float* sesc   = (float*)(ws + B_KT);
    float* out    = (float*)d_out;

    const int trBlocks  = BB * HH * 3 * 8;
    const int pixTiles  = NPIX / 16;             // 4608
    const int convBlocks = 4 * BB * HH * 6;      // 18432 (4-way co blocking)
    const int w3x3n = CC * 9 * CC;               // 589824

    // 0. conversions / repacks
    k_nchw2nhwc<<<trBlocks, 256, 0, stream>>>(x, xt);
    k_repack3x3<<<(w3x3n + 255) / 256, 256, 0, stream>>>(w_ms1, wr1);
    k_repack3x3<<<(w3x3n + 255) / 256, 256, 0, stream>>>(w_ms2, wr2);
    k_repack3x3<<<(w3x3n + 255) / 256, 256, 0, stream>>>(w_ms3, wr3);
    k_repack3x3<<<(w3x3n + 255) / 256, 256, 0, stream>>>(w3, w3r);
    k_cvtbf<<<(CQQ * CC + 255) / 256, 256, 0, stream>>>(wq, wqb, CQQ * CC);
    k_cvtbf<<<(CQQ * CC + 255) / 256, 256, 0, stream>>>(wk, wkb, CQQ * CC);
    k_cvtbf<<<(CC * CC + 255) / 256, 256, 0, stream>>>(wv, wvb, CC * CC);
    k_cvtbf<<<(CC * CC + 255) / 256, 256, 0, stream>>>(w1, w1b, CC * CC);
    // 1. multi-scale conv (WMMA implicit GEMM, 4 co tiles per wave)
    k_msconv<<<convBlocks, 32, 0, stream>>>(xt, wr1, wr2, wr3, b_ms1, b_ms2, b_ms3, msbf);
    // 2. q/k/v projections
    k_qkgemm<<<pixTiles, 32, 0, stream>>>(msbf, wqb, bq, qt);
    k_qkgemm<<<pixTiles, 32, 0, stream>>>(msbf, wkb, bk, kt);
    k_vgemm<<<4 * pixTiles, 32, 0, stream>>>(msbf, wvb, bv, vbf);
    k_vtrans<<<BB * CC * 9, 256, 0, stream>>>(vbf, vt);
    // 3. criss-cross scores + joint softmax
    k_scoreH<<<BB * WW * 6, 32, 0, stream>>>(qt, kt, e);
    k_scoreW<<<BB * HH * 6, 32, 0, stream>>>(qt, kt, e);
    k_softmax192<<<BB * HH * WW, 32, 0, stream>>>(e, att);
    // 4. att x V (4 c tiles per wave)
    k_outH<<<BB * WW * 4 * 6, 32, 0, stream>>>(vt, att, osum);
    k_outW<<<BB * HH * 4 * 6, 32, 0, stream>>>(vbf, att, osum);
    // 5. residual merge, pooling, SE
    k_xr<<<(int)((NTOT + 255) / 256), 256, 0, stream>>>(osum, x, gamma, xr);
    k_nchw2nhwc<<<trBlocks, 256, 0, stream>>>(xr, xrt);
    k_pool<<<BB * CC, 256, 0, stream>>>(xr, pooled);
    k_se<<<BB * CC, 32, 0, stream>>>(pooled, w_se, b_se, sesc);
    // 6. x1 = BN(conv1x1(xr))
    k_x1gemm<<<4 * pixTiles, 32, 0, stream>>>(xrt, w1b, b1, x1, bn1g, bn1b, bn1m, bn1v);
    // 7. final conv3x3 + BN + SE + SiLU
    k_final<<<convBlocks, 32, 0, stream>>>(xrt, w3r, b3, bn3g, bn3b, bn3m, bn3v,
                                           x1, xr, sesc, out);
}
